// GraphTransformer_35605278884481
// MI455X (gfx1250) — matrix operand order repaired and checked by reading, not compiled, not run on hardware
//
#include <hip/hip_runtime.h>
#include <hip/hip_bf16.h>

// ---------------------------------------------------------------------------
// Graph Transformer (B=4, N=256, DIM=128, H=8, DH=64, inner=512, FF=512, L=4)
// Edge-conditioned attention, algebraically refactored to avoid materializing
// e = edges @ We. Edge tensor (134 MB) is L2-resident on MI455X (192 MB L2),
// so the two-pass edge stream is near the traffic floor. All dense algebra on
// v_wmma_f32_16x16x32_bf16 (wave32, 16x64 tile/wave, contiguous b128 loads).
// ---------------------------------------------------------------------------

typedef __attribute__((ext_vector_type(16))) __bf16 v16bf;
typedef __attribute__((ext_vector_type(8)))  float  v8f;

#define BATCH 4
#define NNODE 256
#define DIMC  128
#define HEADS 8
#define DHEAD 64
#define INNER 512
#define ROWS  (BATCH * NNODE)          // 1024
#define DEPTH 4
#define NEG_INF (-3.402823466e38f)

enum { ACT_NONE = 0, ACT_GELU = 1 };

union AliasV16 { uint4 u4[2]; v16bf v; };

__device__ __forceinline__ float gelu_tanh(float v) {
    float c = 0.7978845608028654f * (v + 0.044715f * v * v * v);
    return 0.5f * v * (1.0f + tanhf(c));
}

// ---------------------------------------------------------------------------
// WMMA GEMM: C[z] = act(A_bf16[z] @ B_bf16[z](NK-layout) + bias[z]) (+= if ACCUM)
// A: row-major [M,K] bf16.  B: [N,K] bf16 (K contiguous).  One wave -> 16x64.
// Per-z offsets: off = (z / zdiv) * so + (z % zdiv) * si   (element units)
// Fragment layouts used here (contiguous k order, same for A and B; not the ISA table order):
//   A 16x32: lanes 0-15 rows, K {0..15}; lanes 16-31 K {16..31}
//            => two contiguous 8xbf16 (16B) runs per lane.
//   B 32x16: lane n = col, K kb..kb+15 contiguous => one 32B run per lane.
//   C 16x16: VGPR r: lanes 0-15 -> M=r, lanes 16-31 -> M=r+8, N=lane%16.
// ---------------------------------------------------------------------------
template<int ACT, bool ACCUM, bool TRANSC>
__global__ __launch_bounds__(32)
void wmma_gemm_kernel(const __bf16* __restrict__ A, int ldA, long soA, long siA,
                      const __bf16* __restrict__ Bm, int ldB, long soB, long siB,
                      const float*  __restrict__ bias, long soBias,
                      float* __restrict__ C, int ldC, long soC, long siC,
                      __bf16* __restrict__ Cbf,
                      int K, int zdiv)
{
    const int lane = threadIdx.x;            // wave32
    const int col0 = blockIdx.x * 64;
    const int row0 = blockIdx.y * 16;
    const long zo = blockIdx.z / zdiv, zi = blockIdx.z % zdiv;
    A  += zo * soA + zi * siA;
    Bm += zo * soB + zi * siB;
    C  += zo * soC + zi * siC;
    if (Cbf)  Cbf  += zo * soC + zi * siC;
    if (bias) bias += zo * soBias;

    const int half = (lane >> 4) & 1;
    const int lm   = lane & 15;

    const __bf16* Arow = A + (long)(row0 + lm) * ldA;

    v8f acc[4] = {};

    for (int k0 = 0; k0 < K; k0 += 32) {
        AliasV16 au;
        au.u4[0] = *(const uint4*)(Arow + k0 + 16 * half);
        au.u4[1] = *(const uint4*)(Arow + k0 + 16 * half + 8);
        #pragma unroll
        for (int t = 0; t < 4; ++t) {
            const __bf16* Bp = Bm + (long)(col0 + t * 16 + lm) * ldB + k0 + 16 * half;
            AliasV16 bu;
            bu.u4[0] = *(const uint4*)(Bp);
            bu.u4[1] = *(const uint4*)(Bp + 8);
            acc[t] = __builtin_amdgcn_wmma_f32_16x16x32_bf16(
                         false, au.v, false, bu.v, (short)0, acc[t], false, false);
        }
    }

    #pragma unroll
    for (int t = 0; t < 4; ++t) {
        const int col = col0 + t * 16 + lm;
        const float bv = bias ? bias[col] : 0.0f;
        float cv[8];
        #pragma unroll
        for (int r = 0; r < 8; ++r) {
            const int row = row0 + r + 8 * half;
            float v = acc[t][r] + bv;
            if (ACT == ACT_GELU) v = gelu_tanh(v);
            const long idx = TRANSC ? ((long)col * ldC + row) : ((long)row * ldC + col);
            if (ACCUM) v += C[idx];
            C[idx] = v;
            cv[r] = v;
        }
        if (Cbf) {
            #pragma unroll
            for (int r = 0; r < 8; ++r) {
                const int row = row0 + r + 8 * half;
                const long idx = TRANSC ? ((long)col * ldC + row) : ((long)row * ldC + col);
                Cbf[idx] = (__bf16)cv[r];
            }
        }
    }
}

// ---------------------------------------------------------------------------
// Weight prep: out[z][n*K + k] = (bf16) in[z][k*N + n]   (per-layer z)
// ---------------------------------------------------------------------------
__global__ __launch_bounds__(256)
void transpose_to_bf16_kernel(const float* __restrict__ in,
                              __bf16* __restrict__ out, int K, int N)
{
    const long zoff = (long)blockIdx.z * K * N;
    const int idx = blockIdx.x * 256 + threadIdx.x;
    if (idx < K * N) {
        const int n = idx / K, k = idx - n * K;
        out[zoff + idx] = (__bf16)in[zoff + (long)k * N + n];
    }
}

__global__ void f32_to_bf16_kernel(const float* __restrict__ in,
                                   __bf16* __restrict__ out, int n)
{
    int i = blockIdx.x * blockDim.x + threadIdx.x;
    if (i < n) out[i] = (__bf16)in[i];
}

// ---------------------------------------------------------------------------
// LayerNorm over last dim (128), write bf16 for the following GEMMs.
// ---------------------------------------------------------------------------
__global__ __launch_bounds__(128)
void layernorm_bf16_kernel(const float* __restrict__ x,
                           const float* __restrict__ g,
                           const float* __restrict__ b,
                           __bf16* __restrict__ out)
{
    __shared__ float red[128];
    const int row = blockIdx.x, t = threadIdx.x;
    float v = x[(long)row * DIMC + t];
    red[t] = v; __syncthreads();
    for (int s = 64; s > 0; s >>= 1) { if (t < s) red[t] += red[t + s]; __syncthreads(); }
    float mu = red[0] * (1.0f / DIMC);
    __syncthreads();
    float d = v - mu;
    red[t] = d * d; __syncthreads();
    for (int s = 64; s > 0; s >>= 1) { if (t < s) red[t] += red[t + s]; __syncthreads(); }
    float var = red[0] * (1.0f / DIMC);
    float y = d * rsqrtf(var + 1e-6f) * g[t] + b[t];
    out[(long)row * DIMC + t] = (__bf16)y;
}

// ---------------------------------------------------------------------------
// Attention scores + softmax. Block = (b,i); thread = j.
// sim[h] = ( qk[b,h,i,j] + P_i.edges_ij + q_i.be_h ) * DH^-0.5, mask, softmax.
// qk precomputed by WMMA (Q Kh^T); this kernel just streams the edge rows.
// ---------------------------------------------------------------------------
__global__ __launch_bounds__(256)
void attn_scores_kernel(const float* __restrict__ q,      // [ROWS,512] (qbe only)
                        const float* __restrict__ qk,     // [B,H,N,N]
                        const float* __restrict__ P,      // [ROWS,1024] h*128+c
                        const float* __restrict__ be,     // [512]
                        const float* __restrict__ edges,  // [B,N,N,128]
                        const unsigned char* __restrict__ mask, // [B*N]
                        float* __restrict__ attn,         // [B,H,N,N]
                        __bf16* __restrict__ attn_bf)
{
    const int bi = blockIdx.x;
    const int b = bi >> 8, i = bi & 255;
    const int j = threadIdx.x;
    __shared__ float Ps[HEADS * DIMC];
    __shared__ float qbe[HEADS];
    __shared__ float red[256];

    #pragma unroll
    for (int t = 0; t < 4; ++t)
        Ps[j + 256 * t] = P[(long)bi * 1024 + j + 256 * t];
    if (j < HEADS) {
        float s = 0.f;
        const float* qr = q + (long)bi * INNER + j * DHEAD;
        for (int d = 0; d < DHEAD; ++d) s += qr[d] * be[j * DHEAD + d];
        qbe[j] = s;
    }
    __syncthreads();

    float s[HEADS];
    #pragma unroll
    for (int h = 0; h < HEADS; ++h)
        s[h] = qbe[h] + qk[(((long)(b * HEADS + h) * 256 + i) * 256) + j];

    // P_i . edges_ij (the q.e term, refactored)
    const float4* e4 = reinterpret_cast<const float4*>(edges + ((long)bi * 256 + j) * DIMC);
    for (int c4 = 0; c4 < 32; ++c4) {
        if (c4 + 8 < 32) __builtin_prefetch(&e4[c4 + 8], 0, 0);
        float4 ev = e4[c4];
        #pragma unroll
        for (int h = 0; h < HEADS; ++h) {
            const float* Ph = &Ps[h * DIMC + 4 * c4];
            s[h] += Ph[0] * ev.x + Ph[1] * ev.y + Ph[2] * ev.z + Ph[3] * ev.w;
        }
    }
    const float scale = 0.125f;  // 64^-0.5
    const bool valid = mask[b * 256 + i] && mask[b * 256 + j];
    #pragma unroll
    for (int h = 0; h < HEADS; ++h)
        s[h] = valid ? s[h] * scale : NEG_INF;

    for (int h = 0; h < HEADS; ++h) {
        red[j] = s[h]; __syncthreads();
        for (int st = 128; st > 0; st >>= 1) { if (j < st) red[j] = fmaxf(red[j], red[j + st]); __syncthreads(); }
        float mx = red[0]; __syncthreads();
        float ex = __expf(s[h] - mx);
        red[j] = ex; __syncthreads();
        for (int st = 128; st > 0; st >>= 1) { if (j < st) red[j] += red[j + st]; __syncthreads(); }
        float a = ex / red[0];
        __syncthreads();
        long o = (((long)(b * HEADS + h) * 256 + i) * 256) + j;
        attn[o] = a;
        attn_bf[o] = (__bf16)a;
    }
}

// ---------------------------------------------------------------------------
// agg[b,i,h,:] = sum_j attn[b,h,i,j] * edges[b,i,j,:]  (8 FMA / edge element)
// Block = (b,i), threads over 128 channels; edges read fully coalesced.
// ---------------------------------------------------------------------------
__global__ __launch_bounds__(128)
void attn_edge_agg_kernel(const float* __restrict__ attn,   // [B,H,N,N]
                          const float* __restrict__ edges,  // [B,N,N,128]
                          float* __restrict__ agg,          // [ROWS, H*128]
                          __bf16* __restrict__ agg_bf)
{
    const int bi = blockIdx.x;
    const int b = bi >> 8, i = bi & 255;
    const int c = threadIdx.x;
    __shared__ float aL[HEADS * 256];
    for (int t = c; t < HEADS * 256; t += 128) {
        int h = t >> 8, j = t & 255;
        aL[t] = attn[(((long)(b * HEADS + h) * 256 + i) * 256) + j];
    }
    __syncthreads();
    float acc[HEADS] = {0.f, 0.f, 0.f, 0.f, 0.f, 0.f, 0.f, 0.f};
    const float* erow = edges + (long)bi * 256 * DIMC;
    for (int j = 0; j < 256; ++j) {
        if (j + 4 < 256) __builtin_prefetch(&erow[(j + 4) * DIMC + c], 0, 0);
        float ec = erow[j * DIMC + c];
        #pragma unroll
        for (int h = 0; h < HEADS; ++h) acc[h] += aL[h * 256 + j] * ec;
    }
    #pragma unroll
    for (int h = 0; h < HEADS; ++h) {
        long o = (long)bi * (HEADS * DIMC) + h * DIMC + c;
        agg[o] = acc[h];
        agg_bf[o] = (__bf16)acc[h];
    }
}

// ---------------------------------------------------------------------------
// Gated residual: gate = sigmoid([x, res, x-res] . Wg); out = x*g + res*(1-g)
// ---------------------------------------------------------------------------
__global__ __launch_bounds__(128)
void gated_residual_kernel(const float* __restrict__ xin,
                           const float* __restrict__ res,
                           const float* __restrict__ Wg,   // [384]
                           float* __restrict__ xout)
{
    __shared__ float red[128];
    const int row = blockIdx.x, t = threadIdx.x;
    float a = xin[(long)row * DIMC + t];
    float r = res[(long)row * DIMC + t];
    red[t] = a * Wg[t] + r * Wg[DIMC + t] + (a - r) * Wg[2 * DIMC + t];
    __syncthreads();
    for (int s = 64; s > 0; s >>= 1) { if (t < s) red[t] += red[t + s]; __syncthreads(); }
    float gate = 1.0f / (1.0f + __expf(-red[0]));
    xout[(long)row * DIMC + t] = a * gate + r * (1.0f - gate);
}

__global__ void copy_f32_kernel(const float* __restrict__ in,
                                float* __restrict__ out, int n)
{
    int i = blockIdx.x * blockDim.x + threadIdx.x;
    if (i < n) out[i] = in[i];
}

// ---------------------------------------------------------------------------
extern "C" void kernel_launch(void* const* d_in, const int* in_sizes, int n_in,
                              void* d_out, int out_size, void* d_ws, size_t ws_size,
                              hipStream_t stream)
{
    const float* nodes = (const float*)d_in[0];
    const float* edges = (const float*)d_in[1];
    const float* ln1_g = (const float*)d_in[2];
    const float* ln1_b = (const float*)d_in[3];
    const float* Wq = (const float*)d_in[4];  const float* bq = (const float*)d_in[5];
    const float* Wk = (const float*)d_in[6];  const float* bk = (const float*)d_in[7];
    const float* Wv = (const float*)d_in[8];  const float* bv = (const float*)d_in[9];
    const float* We = (const float*)d_in[10]; const float* be = (const float*)d_in[11];
    const float* Wo = (const float*)d_in[12]; const float* bo = (const float*)d_in[13];
    const float* Wg1 = (const float*)d_in[14];
    const float* ln2_g = (const float*)d_in[15];
    const float* ln2_b = (const float*)d_in[16];
    const float* W1 = (const float*)d_in[17]; const float* b1 = (const float*)d_in[18];
    const float* W2 = (const float*)d_in[19]; const float* b2 = (const float*)d_in[20];
    const float* Wg2 = (const float*)d_in[21];
    const unsigned char* mask = (const unsigned char*)d_in[22];

    char* base = (char*)d_ws;
    size_t off = 0;
    auto alloc = [&](size_t bytes) -> void* {
        void* p = base + off;
        off += (bytes + 255) & ~(size_t)255;
        return p;
    };
    const size_t WSZ = (size_t)DIMC * INNER;       // 65536 elements per layer
    // bf16 weight copies (K-contiguous [N,K] layouts), all DEPTH layers
    __bf16* Wq_t = (__bf16*)alloc(DEPTH * WSZ * 2);
    __bf16* Wk_t = (__bf16*)alloc(DEPTH * WSZ * 2);
    __bf16* Wv_t = (__bf16*)alloc(DEPTH * WSZ * 2);
    __bf16* We_t = (__bf16*)alloc(DEPTH * WSZ * 2);   // [512,128]
    __bf16* We_b = (__bf16*)alloc(DEPTH * WSZ * 2);   // plain bf16 [128,512]
    __bf16* Wo_t = (__bf16*)alloc(DEPTH * WSZ * 2);   // [128,512]
    __bf16* W1_t = (__bf16*)alloc(DEPTH * WSZ * 2);   // [512,128]
    __bf16* W2_t = (__bf16*)alloc(DEPTH * WSZ * 2);   // [128,512]
    // activations
    __bf16* xn_bf    = (__bf16*)alloc((size_t)ROWS * DIMC * 2);
    float*  q_f32    = (float*)alloc((size_t)ROWS * INNER * 4);
    float*  k_f32    = (float*)alloc((size_t)ROWS * INNER * 4);
    __bf16* q_bf     = (__bf16*)alloc((size_t)ROWS * INNER * 2);
    __bf16* k_bf     = (__bf16*)alloc((size_t)ROWS * INNER * 2);
    float*  vt_f32   = (float*)alloc((size_t)ROWS * INNER * 4);  // [B,512,256]
    __bf16* vt_bf    = (__bf16*)alloc((size_t)ROWS * INNER * 2); // [B,512,256]
    float*  P_f32    = (float*)alloc((size_t)ROWS * HEADS * DIMC * 4);
    float*  qk_f32   = (float*)alloc((size_t)BATCH * HEADS * NNODE * NNODE * 4);
    float*  attn_f32 = (float*)alloc((size_t)BATCH * HEADS * NNODE * NNODE * 4);
    __bf16* attn_bf  = (__bf16*)alloc((size_t)BATCH * HEADS * NNODE * NNODE * 2);
    float*  agg_f32  = (float*)alloc((size_t)ROWS * HEADS * DIMC * 4);
    __bf16* agg_bf   = (__bf16*)alloc((size_t)ROWS * HEADS * DIMC * 2);
    float*  ocat     = (float*)alloc((size_t)ROWS * INNER * 4);
    __bf16* ocat_bf  = (__bf16*)alloc((size_t)ROWS * INNER * 2);
    float*  attn_out = (float*)alloc((size_t)ROWS * DIMC * 4);
    float*  xmid     = (float*)alloc((size_t)ROWS * DIMC * 4);
    float*  xbuf     = (float*)alloc((size_t)ROWS * DIMC * 4);
    float*  hid      = (float*)alloc((size_t)ROWS * INNER * 4);
    __bf16* hid_bf   = (__bf16*)alloc((size_t)ROWS * INNER * 2);
    float*  ff_out   = (float*)alloc((size_t)ROWS * DIMC * 4);

    // ---- weight prep (once per launch) ----
    {
        const dim3 gt((int)((WSZ + 255) / 256), 1, DEPTH);
        transpose_to_bf16_kernel<<<gt, 256, 0, stream>>>(Wq, Wq_t, DIMC, INNER);
        transpose_to_bf16_kernel<<<gt, 256, 0, stream>>>(Wk, Wk_t, DIMC, INNER);
        transpose_to_bf16_kernel<<<gt, 256, 0, stream>>>(Wv, Wv_t, DIMC, INNER);
        transpose_to_bf16_kernel<<<gt, 256, 0, stream>>>(We, We_t, DIMC, INNER);
        transpose_to_bf16_kernel<<<gt, 256, 0, stream>>>(Wo, Wo_t, INNER, DIMC);
        transpose_to_bf16_kernel<<<gt, 256, 0, stream>>>(W1, W1_t, DIMC, INNER);
        transpose_to_bf16_kernel<<<gt, 256, 0, stream>>>(W2, W2_t, INNER, DIMC);
        f32_to_bf16_kernel<<<(int)((DEPTH * WSZ + 255) / 256), 256, 0, stream>>>(
            We, We_b, (int)(DEPTH * WSZ));
    }

    const dim3 g_qk(INNER / 64, ROWS / 16, 1);
    const dim3 g_v(INNER / 64, NNODE / 16, BATCH);
    const dim3 g_P(DIMC / 64, ROWS / 16, HEADS);
    const dim3 g_sim(NNODE / 64, NNODE / 16, BATCH * HEADS);
    const dim3 g_av(1, NNODE / 16, BATCH * HEADS);
    const dim3 g_oe(1, ROWS / 16, HEADS);
    const dim3 g_wo(DIMC / 64, ROWS / 16, 1);
    const dim3 g_f1(INNER / 64, ROWS / 16, 1);
    const dim3 g_f2(DIMC / 64, ROWS / 16, 1);

    for (int l = 0; l < DEPTH; ++l) {
        const float* x_in = (l == 0) ? nodes : xbuf;
        const __bf16* Wq_l = Wq_t + (size_t)l * WSZ;
        const __bf16* Wk_l = Wk_t + (size_t)l * WSZ;
        const __bf16* Wv_l = Wv_t + (size_t)l * WSZ;
        const __bf16* Wet_l = We_t + (size_t)l * WSZ;
        const __bf16* Web_l = We_b + (size_t)l * WSZ;
        const __bf16* Wo_l = Wo_t + (size_t)l * WSZ;
        const __bf16* W1_l = W1_t + (size_t)l * WSZ;
        const __bf16* W2_l = W2_t + (size_t)l * WSZ;
        const float* bq_l = bq + (size_t)l * INNER;
        const float* bk_l = bk + (size_t)l * INNER;
        const float* bv_l = bv + (size_t)l * INNER;
        const float* be_l = be + (size_t)l * INNER;
        const float* bo_l = bo + (size_t)l * DIMC;
        const float* b1_l = b1 + (size_t)l * INNER;
        const float* b2_l = b2 + (size_t)l * DIMC;

        // --- PreNorm 1 ---
        layernorm_bf16_kernel<<<ROWS, 128, 0, stream>>>(
            x_in, ln1_g + l * DIMC, ln1_b + l * DIMC, xn_bf);

        // --- Q, K projections (WMMA; both keep bf16 copies) ---
        wmma_gemm_kernel<ACT_NONE, false, false><<<g_qk, 32, 0, stream>>>(
            xn_bf, DIMC, 0, 0, Wq_l, DIMC, 0, 0, bq_l, 0,
            q_f32, INNER, 0, 0, q_bf, DIMC, 1);
        wmma_gemm_kernel<ACT_NONE, false, false><<<g_qk, 32, 0, stream>>>(
            xn_bf, DIMC, 0, 0, Wk_l, DIMC, 0, 0, bk_l, 0,
            k_f32, INNER, 0, 0, k_bf, DIMC, 1);
        // --- V projection, stored transposed [B,512,256] (WMMA, TRANSC) ---
        wmma_gemm_kernel<ACT_NONE, false, true><<<g_v, 32, 0, stream>>>(
            xn_bf, DIMC, (long)NNODE * DIMC, 0, Wv_l, DIMC, 0, 0, bv_l, 0,
            vt_f32, NNODE, (long)INNER * NNODE, 0, vt_bf, DIMC, 1);

        // --- P = Q_h @ We_h^T per head (WMMA; B = We bf16 rows, K-contig) ---
        wmma_gemm_kernel<ACT_NONE, false, false><<<g_P, 32, 0, stream>>>(
            q_bf, INNER, DHEAD, 0, Web_l, INNER, DHEAD, 0, nullptr, 0,
            P_f32, HEADS * DIMC, DIMC, 0, nullptr, DHEAD, 1);

        // --- qk = Q_h K_h^T per (b,h) (WMMA) ---
        wmma_gemm_kernel<ACT_NONE, false, false><<<g_sim, 32, 0, stream>>>(
            q_bf, INNER, (long)NNODE * INNER, DHEAD,
            k_bf, INNER, (long)NNODE * INNER, DHEAD, nullptr, 0,
            qk_f32, NNODE, (long)HEADS * NNODE * NNODE, (long)NNODE * NNODE, nullptr,
            DHEAD, HEADS);

        // --- scores + softmax (edge stream pass 1) ---
        attn_scores_kernel<<<ROWS, 256, 0, stream>>>(
            q_f32, qk_f32, P_f32, be_l, edges, mask, attn_f32, attn_bf);

        // --- out_v = attn @ V per (b,h) (WMMA; B = V^T, K-contig) ---
        wmma_gemm_kernel<ACT_NONE, false, false><<<g_av, 32, 0, stream>>>(
            attn_bf, NNODE, (long)HEADS * NNODE * NNODE, (long)NNODE * NNODE,
            vt_bf, NNODE, (long)INNER * NNODE, (long)DHEAD * NNODE, nullptr, 0,
            ocat, INNER, (long)NNODE * INNER, DHEAD, nullptr,
            NNODE, HEADS);

        // --- agg = attn-weighted edge rows (edge stream pass 2) ---
        attn_edge_agg_kernel<<<ROWS, 128, 0, stream>>>(
            attn_f32, edges, agg_f32, agg_bf);

        // --- out_e = agg_h @ We_h + be_h, accumulated into out_v (WMMA) ---
        wmma_gemm_kernel<ACT_NONE, true, false><<<g_oe, 32, 0, stream>>>(
            agg_bf, HEADS * DIMC, DIMC, 0, Wet_l, DIMC, (long)DHEAD * DIMC, 0,
            be_l, DHEAD,
            ocat, INNER, DHEAD, 0, nullptr, DIMC, 1);

        f32_to_bf16_kernel<<<(ROWS * INNER + 255) / 256, 256, 0, stream>>>(
            ocat, ocat_bf, ROWS * INNER);

        // --- output projection Wo (WMMA) ---
        wmma_gemm_kernel<ACT_NONE, false, false><<<g_wo, 32, 0, stream>>>(
            ocat_bf, INNER, 0, 0, Wo_l, INNER, 0, 0, bo_l, 0,
            attn_out, DIMC, 0, 0, nullptr, INNER, 1);

        // --- gated residual 1 ---
        gated_residual_kernel<<<ROWS, 128, 0, stream>>>(
            attn_out, x_in, Wg1 + (size_t)l * 3 * DIMC, xmid);

        // --- PreNorm 2 + FFN ---
        layernorm_bf16_kernel<<<ROWS, 128, 0, stream>>>(
            xmid, ln2_g + l * DIMC, ln2_b + l * DIMC, xn_bf);
        wmma_gemm_kernel<ACT_GELU, false, false><<<g_f1, 32, 0, stream>>>(
            xn_bf, DIMC, 0, 0, W1_l, DIMC, 0, 0, b1_l, 0,
            hid, INNER, 0, 0, hid_bf, DIMC, 1);
        wmma_gemm_kernel<ACT_NONE, false, false><<<g_f2, 32, 0, stream>>>(
            hid_bf, INNER, 0, 0, W2_l, INNER, 0, 0, b2_l, 0,
            ff_out, DIMC, 0, 0, nullptr, INNER, 1);

        // --- gated residual 2 ---
        gated_residual_kernel<<<ROWS, 128, 0, stream>>>(
            ff_out, xmid, Wg2 + (size_t)l * 3 * DIMC, xbuf);
    }

    copy_f32_kernel<<<(ROWS * DIMC + 255) / 256, 256, 0, stream>>>(
        xbuf, (float*)d_out, ROWS * DIMC);
}
